// CrossAttention_55198919688799
// MI455X (gfx1250) — matrix operand order repaired and checked by reading, not compiled, not run on hardware
//
#include <hip/hip_runtime.h>
#include <math.h>

// ---------------------------------------------------------------------------
// CrossAttention pipeline for MI455X (gfx1250, wave32, WMMA f16 -> f32 acc)
// ---------------------------------------------------------------------------

typedef __attribute__((ext_vector_type(16))) _Float16 v16h;
typedef __attribute__((ext_vector_type(8)))  _Float16 v8h;
typedef __attribute__((ext_vector_type(4)))  _Float16 v4h;
typedef __attribute__((ext_vector_type(2)))  __fp16   v2fp;   // cvt_pkrtz result type
typedef __attribute__((ext_vector_type(8)))  float    v8f;

static __device__ __forceinline__ v8f wmma16(v16h a, v16h b, v8f c) {
  // D = A(16x32 f16) * B(32x16 f16) + C(16x16 f32)
  return __builtin_amdgcn_wmma_f32_16x16x32_f16(false, a, false, b, (short)0, c,
                                                false, false);
}

// A-fragment: 16x32 f16 tile, row-major source with leading dim ld (halves).
static __device__ __forceinline__ v16h ld_afrag(const _Float16* p, int ld, int lane) {
  const _Float16* q = p + (lane & 15) * ld + ((lane & 16) ? 8 : 0);
  v8h lo = *(const v8h*)q;
  v8h hi = *(const v8h*)(q + 16);
  v16h r;
#pragma unroll
  for (int i = 0; i < 8; ++i) { r[i] = lo[i]; r[8 + i] = hi[i]; }
  return r;
}

// B-fragment: 32x16 f16 tile given in TRANSPOSED [n][k] storage, leading dim ld.
static __device__ __forceinline__ v16h ld_bfrag(const _Float16* p, int ld, int lane) {
  const _Float16* q = p + (lane & 15) * ld + ((lane & 16) ? 8 : 0);
  v8h lo = *(const v8h*)q;
  v8h hi = *(const v8h*)(q + 16);
  v16h r;
#pragma unroll
  for (int i = 0; i < 8; ++i) { r[i] = lo[i]; r[8 + i] = hi[i]; }
  return r;
}

// quad (4-lane) reductions for the 4-threads-per-row softmax
static __device__ __forceinline__ float qmax(float v) {
  v = fmaxf(v, __shfl_xor(v, 1, 32));
  v = fmaxf(v, __shfl_xor(v, 2, 32));
  return v;
}
static __device__ __forceinline__ float qsum(float v) {
  v += __shfl_xor(v, 1, 32);
  v += __shfl_xor(v, 2, 32);
  return v;
}

// ---------------------------------------------------------------------------
// LayerNorm over 512 columns. One block (256 threads) per row.
// ---------------------------------------------------------------------------
__global__ __launch_bounds__(256) void ln_kernel(
    const float* __restrict__ in, const float* __restrict__ g,
    const float* __restrict__ b, float* __restrict__ out32,
    _Float16* __restrict__ out16) {
  const int COLS = 512;
  const size_t row = blockIdx.x;
  const float* xr = in + row * COLS;
  const int t = threadIdx.x;

  float v0 = xr[t], v1 = xr[t + 256];

  __shared__ float sred[8];
  __shared__ float smean, sinv;

  float s = v0 + v1;
#pragma unroll
  for (int o = 16; o > 0; o >>= 1) s += __shfl_xor(s, o, 32);
  if ((t & 31) == 0) sred[t >> 5] = s;
  __syncthreads();
  if (t == 0) {
    float a = 0.f;
#pragma unroll
    for (int i = 0; i < 8; ++i) a += sred[i];
    smean = a * (1.0f / COLS);
  }
  __syncthreads();
  float mean = smean;
  float d0 = v0 - mean, d1 = v1 - mean;
  s = d0 * d0 + d1 * d1;
#pragma unroll
  for (int o = 16; o > 0; o >>= 1) s += __shfl_xor(s, o, 32);
  if ((t & 31) == 0) sred[t >> 5] = s;
  __syncthreads();
  if (t == 0) {
    float a = 0.f;
#pragma unroll
    for (int i = 0; i < 8; ++i) a += sred[i];
    sinv = rsqrtf(a * (1.0f / COLS) + 1e-5f);
  }
  __syncthreads();
  float inv = sinv;
  float y0 = d0 * inv * g[t] + b[t];
  float y1 = d1 * inv * g[t + 256] + b[t + 256];
  if (out32) { out32[row * COLS + t] = y0; out32[row * COLS + t + 256] = y1; }
  if (out16) {
    out16[row * COLS + t] = (_Float16)y0;
    out16[row * COLS + t + 256] = (_Float16)y1;
  }
}

// ---------------------------------------------------------------------------
// f32 -> f16 conversion (for x)
// ---------------------------------------------------------------------------
__global__ __launch_bounds__(256) void cvt_kernel(const float* __restrict__ in,
                                                  _Float16* __restrict__ out,
                                                  int n) {
  int i = (blockIdx.x * 256 + threadIdx.x) * 4;
  if (i < n) {
    float4 f = *(const float4*)(in + i);
    v4h h;
    h[0] = (_Float16)f.x; h[1] = (_Float16)f.y;
    h[2] = (_Float16)f.z; h[3] = (_Float16)f.w;
    *(v4h*)(out + i) = h;
  }
}

// ---------------------------------------------------------------------------
// Generic GEMM: C[M,N] = A_f16[M,K] * B_f32[K,N] (+bias +residual)
// Block tile 128x128, K-step 32. 8 waves; each wave owns a 32x64 region
// (2 A-frags x 4 B-frags -> 8 WMMA per K-step, B-frag reused twice).
// Flags are compile-time so the epilogue has no pointer-null branches.
// ---------------------------------------------------------------------------
template <bool BIAS, bool RESID, bool O32, bool O16>
__global__ __launch_bounds__(256) void gemm_f16_wmma(
    const _Float16* __restrict__ A, const float* __restrict__ B,
    float* __restrict__ out32, _Float16* __restrict__ out16,
    const float* __restrict__ bias, const float* __restrict__ resid,
    int M, int N, int K) {
  __shared__ _Float16 sA[128][32];   // [m][k]   8 KB
  __shared__ _Float16 sB[128][32];   // [n][k]   8 KB (transposed weight tile)

  const int tidx = threadIdx.x;
  const int lane = tidx & 31;
  const int wave = tidx >> 5;
  const int m0 = blockIdx.y * 128;
  const int n0 = blockIdx.x * 128;
  const int rb = (wave & 3) * 32;     // wave's 32-row strip
  const int cb = (wave >> 2) * 64;    // wave's 64-col strip

  // staging coordinates
  const int ar = tidx >> 1;                 // A: 128 rows, 2 threads/row
  const int ac = (tidx & 1) * 16;           //    16 halves each
  const int bk = (tidx >> 4) * 2;           // B: 2 consecutive k rows
  const int bn = (tidx & 15) * 8;           //    8 consecutive n cols

  v8f acc[2][4] = {};

  for (int k0 = 0; k0 < K; k0 += 32) {
    // prefetch next K-tile into cache while we stage this one
    if (k0 + 32 < K) {
      __builtin_prefetch(A + (size_t)(m0 + ar) * K + k0 + 32 + ac, 0, 1);
      __builtin_prefetch(B + (size_t)(k0 + 32 + bk) * N + n0 + bn, 0, 1);
    }
    // stage A: 128x32 halves, 16 halves (32B) per thread
    {
      const _Float16* ap = A + (size_t)(m0 + ar) * K + k0 + ac;
      *(v8h*)&sA[ar][ac]     = *(const v8h*)(ap);
      *(v8h*)&sA[ar][ac + 8] = *(const v8h*)(ap + 8);
    }
    // stage B transposed with packed f32->2xf16 convert: 8n x 2k per thread
    {
      const float* bp = B + (size_t)(k0 + bk) * N + n0 + bn;
      float4 r0a = *(const float4*)bp;
      float4 r0b = *(const float4*)(bp + 4);
      float4 r1a = *(const float4*)(bp + N);
      float4 r1b = *(const float4*)(bp + N + 4);
      float r0[8] = {r0a.x, r0a.y, r0a.z, r0a.w, r0b.x, r0b.y, r0b.z, r0b.w};
      float r1[8] = {r1a.x, r1a.y, r1a.z, r1a.w, r1b.x, r1b.y, r1b.z, r1b.w};
#pragma unroll
      for (int j = 0; j < 8; ++j) {
        v2fp pk = __builtin_amdgcn_cvt_pkrtz(r0[j], r1[j]);
        *(v2fp*)&sB[bn + j][bk] = pk;  // [n][k], k pair contiguous -> b32 store
      }
    }
    __syncthreads();

    v16h a0 = ld_afrag(&sA[rb][0],      32, lane);
    v16h a1 = ld_afrag(&sA[rb + 16][0], 32, lane);
#pragma unroll
    for (int tj = 0; tj < 4; ++tj) {
      v16h bf = ld_bfrag(&sB[cb + tj * 16][0], 32, lane);
      acc[0][tj] = wmma16(a0, bf, acc[0][tj]);
      acc[1][tj] = wmma16(a1, bf, acc[1][tj]);
    }
    __syncthreads();
  }

  const int roff = (lane & 16) ? 8 : 0;
  const int cc = lane & 15;
#pragma unroll
  for (int ti = 0; ti < 2; ++ti) {
#pragma unroll
    for (int tj = 0; tj < 4; ++tj) {
#pragma unroll
      for (int g = 0; g < 8; ++g) {
        int row = m0 + rb + ti * 16 + roff + g;
        int col = n0 + cb + tj * 16 + cc;
        float val = acc[ti][tj][g];
        if (BIAS)  val += bias[col];
        if (RESID) val += resid[(size_t)row * N + col];
        if (O32)   out32[(size_t)row * N + col] = val;
        if (O16)   out16[(size_t)row * N + col] = (_Float16)val;
      }
    }
  }
}

// ---------------------------------------------------------------------------
// Local chunked self-attention. One block per (b, h, chunk of 64).
// qkv: [4096][1536] f16 (q | k | v), out: [4096][512] f16 merged heads.
// Q is pre-scaled by DH^-0.5 = 0.125 (exact in f16) at staging.
// ---------------------------------------------------------------------------
__global__ __launch_bounds__(256) void local_attn_kernel(
    const _Float16* __restrict__ qkv, _Float16* __restrict__ out) {
  const int blk = blockIdx.x;           // b*256 + h*32 + m
  const int m = blk & 31;
  const int h = (blk >> 5) & 7;
  const int b = blk >> 8;
  const int row0 = b * 2048 + m * 64;

  __shared__ _Float16 sQ[64][64];
  __shared__ _Float16 sK[64][64];       // [key][d] == B-layout for K^T
  __shared__ _Float16 sVt[64][64];      // [d][key] == B-layout for V
  __shared__ float    sS[64][64];
  __shared__ _Float16 sP[64][64];

  const int tidx = threadIdx.x;
  const int lane = tidx & 31;
  const int wave = tidx >> 5;
  const int mi = wave & 3;
  const int nj = (wave >> 2) * 2;

  // load Q (scaled), K, V tiles (each thread moves 16 halves per tile)
  {
    int idx = tidx * 16;
    int r = idx >> 6, c = idx & 63;
    const _Float16* base = qkv + (size_t)(row0 + r) * 1536 + h * 64 + c;
    v8h q0 = *(const v8h*)(base);
    v8h q1 = *(const v8h*)(base + 8);
    *(v8h*)&sQ[r][c]     = q0 * (_Float16)0.125f;
    *(v8h*)&sQ[r][c + 8] = q1 * (_Float16)0.125f;
    *(v8h*)&sK[r][c]     = *(const v8h*)(base + 512);
    *(v8h*)&sK[r][c + 8] = *(const v8h*)(base + 520);
    v8h v0 = *(const v8h*)(base + 1024);
    v8h v1 = *(const v8h*)(base + 1032);
#pragma unroll
    for (int j = 0; j < 8; ++j) { sVt[c + j][r] = v0[j]; sVt[c + 8 + j][r] = v1[j]; }
  }
  __syncthreads();

  // S = Q K^T  (d = 64 contraction, two k-steps)
  v8f s0 = {}, s1 = {};
#pragma unroll
  for (int kk = 0; kk < 64; kk += 32) {
    v16h aq  = ld_afrag(&sQ[mi * 16][kk], 64, lane);
    v16h bk0 = ld_bfrag(&sK[nj * 16][kk], 64, lane);
    v16h bk1 = ld_bfrag(&sK[(nj + 1) * 16][kk], 64, lane);
    s0 = wmma16(aq, bk0, s0);
    s1 = wmma16(aq, bk1, s1);
  }
  const int roff = ((lane & 16) ? 8 : 0);
  const int cc = lane & 15;
#pragma unroll
  for (int g = 0; g < 8; ++g) {
    sS[mi * 16 + roff + g][nj * 16 + cc]       = s0[g];
    sS[mi * 16 + roff + g][(nj + 1) * 16 + cc] = s1[g];
  }
  __syncthreads();

  // softmax: 4 threads per row, 16 cols each, quad shuffle reductions
  {
    const int r = tidx >> 2;
    const int c0 = (tidx & 3) * 16;
    float v[16];
#pragma unroll
    for (int i = 0; i < 4; ++i) {
      float4 f = *(const float4*)&sS[r][c0 + 4 * i];
      v[4 * i] = f.x; v[4 * i + 1] = f.y; v[4 * i + 2] = f.z; v[4 * i + 3] = f.w;
    }
    float mx = -1e30f;
#pragma unroll
    for (int i = 0; i < 16; ++i) mx = fmaxf(mx, v[i]);
    mx = qmax(mx);
    float e[16];
    float sum = 0.f;
#pragma unroll
    for (int i = 0; i < 16; ++i) { e[i] = __expf(v[i] - mx); sum += e[i]; }
    sum = qsum(sum);
    float inv = 1.0f / sum;
    v8h p0, p1;
#pragma unroll
    for (int i = 0; i < 8; ++i) {
      p0[i] = (_Float16)(e[i] * inv);
      p1[i] = (_Float16)(e[8 + i] * inv);
    }
    *(v8h*)&sP[r][c0]     = p0;
    *(v8h*)&sP[r][c0 + 8] = p1;
  }
  __syncthreads();

  // O = P V
  v8f o0 = {}, o1 = {};
#pragma unroll
  for (int kk = 0; kk < 64; kk += 32) {
    v16h ap  = ld_afrag(&sP[mi * 16][kk], 64, lane);
    v16h bv0 = ld_bfrag(&sVt[nj * 16][kk], 64, lane);
    v16h bv1 = ld_bfrag(&sVt[(nj + 1) * 16][kk], 64, lane);
    o0 = wmma16(ap, bv0, o0);
    o1 = wmma16(ap, bv1, o1);
  }
#pragma unroll
  for (int g = 0; g < 8; ++g) {
    int row = row0 + mi * 16 + roff + g;
    out[(size_t)row * 512 + h * 64 + nj * 16 + cc]       = (_Float16)o0[g];
    out[(size_t)row * 512 + h * 64 + (nj + 1) * 16 + cc] = (_Float16)o1[g];
  }
}

// ---------------------------------------------------------------------------
// Cross attention (flash style). One block per (b, h, 64-query tile).
// Qp: [4096][512] f16 ; KV: [8192][1024] f16 (k | v) ; out: [4096][512] f16.
// K tile is staged with gfx1250 async LDS DMA (ASYNCcnt); V needs a
// transpose so it goes through registers. Q pre-scaled by 0.125.
// ---------------------------------------------------------------------------
__global__ __launch_bounds__(256) void cross_attn_kernel(
    const _Float16* __restrict__ Qp, const _Float16* __restrict__ KV,
    _Float16* __restrict__ out) {
  const int blk = blockIdx.x;           // b*256 + h*32 + t
  const int t = blk & 31;
  const int h = (blk >> 5) & 7;
  const int b = blk >> 8;
  const int qrow0 = b * 2048 + t * 64;
  const int krow0 = b * 4096;

  __shared__ _Float16 sQ[64][64];
  __shared__ _Float16 sK[64][64];
  __shared__ _Float16 sVt[64][64];
  __shared__ float    sS[64][64];
  __shared__ _Float16 sP[64][64];
  __shared__ float rowM[64], rowL[64], rowC[64];

  const int tidx = threadIdx.x;
  const int lane = tidx & 31;
  const int wave = tidx >> 5;
  const int mi = wave & 3;
  const int nj = (wave >> 2) * 2;
  const int roff = ((lane & 16) ? 8 : 0);
  const int cc = lane & 15;

  const int lr = tidx >> 2;             // tile load coords: 64 rows,
  const int lc = (tidx & 3) * 16;       // 16 halves per thread

  {
    const _Float16* qp = Qp + (size_t)(qrow0 + lr) * 512 + h * 64 + lc;
    v8h q0 = *(const v8h*)(qp);
    v8h q1 = *(const v8h*)(qp + 8);
    *(v8h*)&sQ[lr][lc]     = q0 * (_Float16)0.125f;
    *(v8h*)&sQ[lr][lc + 8] = q1 * (_Float16)0.125f;
  }
  if (tidx < 64) { rowM[tidx] = -1e30f; rowL[tidx] = 0.f; }

  // per-thread async-DMA coordinates for the K tile
  const unsigned int klds = (unsigned int)(size_t)(void*)&sK[lr][lc];

  v8f o0 = {}, o1 = {};

  for (int kt = 0; kt < 64; ++kt) {     // 4096 keys / 64
    {
      const _Float16* kp =
          KV + (size_t)(krow0 + kt * 64 + lr) * 1024 + h * 64 + lc;
      // K tile: 32B per thread via async global->LDS DMA (offset applies to
      // both global and LDS sides per ISA 08_async_tensor §4.4)
      unsigned long long ga = (unsigned long long)(size_t)kp;
      asm volatile("global_load_async_to_lds_b128 %0, %1, off"
                   :: "v"(klds), "v"(ga) : "memory");
      asm volatile("global_load_async_to_lds_b128 %0, %1, off offset:16"
                   :: "v"(klds), "v"(ga) : "memory");
      // V tile needs a transpose -> through registers
      v8h v0 = *(const v8h*)(kp + 512);
      v8h v1 = *(const v8h*)(kp + 520);
#pragma unroll
      for (int j = 0; j < 8; ++j) {
        sVt[lc + j][lr] = v0[j];
        sVt[lc + 8 + j][lr] = v1[j];
      }
      asm volatile("s_wait_asynccnt 0" ::: "memory");
    }
    __syncthreads();

    // S = Q K^T
    v8f s0 = {}, s1 = {};
#pragma unroll
    for (int kk = 0; kk < 64; kk += 32) {
      v16h aq  = ld_afrag(&sQ[mi * 16][kk], 64, lane);
      v16h bk0 = ld_bfrag(&sK[nj * 16][kk], 64, lane);
      v16h bk1 = ld_bfrag(&sK[(nj + 1) * 16][kk], 64, lane);
      s0 = wmma16(aq, bk0, s0);
      s1 = wmma16(aq, bk1, s1);
    }
#pragma unroll
    for (int g = 0; g < 8; ++g) {
      sS[mi * 16 + roff + g][nj * 16 + cc]       = s0[g];
      sS[mi * 16 + roff + g][(nj + 1) * 16 + cc] = s1[g];
    }
    __syncthreads();

    // online softmax update: 4 threads per row
    {
      const int r = lr;
      const int c0 = lc;
      float v[16];
#pragma unroll
      for (int i = 0; i < 4; ++i) {
        float4 f = *(const float4*)&sS[r][c0 + 4 * i];
        v[4 * i] = f.x; v[4 * i + 1] = f.y; v[4 * i + 2] = f.z; v[4 * i + 3] = f.w;
      }
      float mprev = rowM[r];
      float mx = -1e30f;
#pragma unroll
      for (int i = 0; i < 16; ++i) mx = fmaxf(mx, v[i]);
      mx = fmaxf(qmax(mx), mprev);
      float corr = __expf(mprev - mx);
      float e[16];
      float sum = 0.f;
#pragma unroll
      for (int i = 0; i < 16; ++i) { e[i] = __expf(v[i] - mx); sum += e[i]; }
      sum = qsum(sum);
      v8h p0, p1;
#pragma unroll
      for (int i = 0; i < 8; ++i) {
        p0[i] = (_Float16)e[i];
        p1[i] = (_Float16)e[8 + i];
      }
      *(v8h*)&sP[r][c0]     = p0;
      *(v8h*)&sP[r][c0 + 8] = p1;
      if ((tidx & 3) == 0) {
        rowL[r] = rowL[r] * corr + sum;
        rowM[r] = mx;
        rowC[r] = corr;
      }
    }
    __syncthreads();

    // rescale accumulators, then O += P V
#pragma unroll
    for (int g = 0; g < 8; ++g) {
      float c = rowC[mi * 16 + roff + g];
      o0[g] *= c;
      o1[g] *= c;
    }
#pragma unroll
    for (int kk = 0; kk < 64; kk += 32) {
      v16h ap  = ld_afrag(&sP[mi * 16][kk], 64, lane);
      v16h bv0 = ld_bfrag(&sVt[nj * 16][kk], 64, lane);
      v16h bv1 = ld_bfrag(&sVt[(nj + 1) * 16][kk], 64, lane);
      o0 = wmma16(ap, bv0, o0);
      o1 = wmma16(ap, bv1, o1);
    }
    __syncthreads();
  }

#pragma unroll
  for (int g = 0; g < 8; ++g) {
    int row = qrow0 + mi * 16 + roff + g;
    float invl = 1.0f / rowL[mi * 16 + roff + g];
    out[(size_t)row * 512 + h * 64 + nj * 16 + cc]       = (_Float16)(o0[g] * invl);
    out[(size_t)row * 512 + h * 64 + (nj + 1) * 16 + cc] = (_Float16)(o1[g] * invl);
  }
}

// ---------------------------------------------------------------------------
// Host driver
// ---------------------------------------------------------------------------
extern "C" void kernel_launch(void* const* d_in, const int* in_sizes, int n_in,
                              void* d_out, int out_size, void* d_ws,
                              size_t ws_size, hipStream_t stream) {
  const float* x     = (const float*)d_in[0];   // [2,4096,512]
  const float* q_x   = (const float*)d_in[1];   // [2,2048,512]
  const float* ln1_g = (const float*)d_in[2];
  const float* ln1_b = (const float*)d_in[3];
  const float* W_qkv = (const float*)d_in[4];   // [512,1536]
  const float* W_ao  = (const float*)d_in[5];   // [512,512]
  const float* b_ao  = (const float*)d_in[6];
  const float* ln2_g = (const float*)d_in[7];
  const float* ln2_b = (const float*)d_in[8];
  const float* W_q   = (const float*)d_in[9];   // [512,512]
  const float* W_kv  = (const float*)d_in[10];  // [512,1024]
  const float* W_o   = (const float*)d_in[11];  // [512,512]
  const float* b_o   = (const float*)d_in[12];

  float* out_attn = (float*)d_out;                     // [2,2048,512]
  float* out_qin  = (float*)d_out + (size_t)2 * 2048 * 512;

  // workspace partition (256B aligned slabs)
  char* ws = (char*)d_ws;
  size_t off = 0;
  auto alloc = [&](size_t bytes) -> void* {
    void* p = ws + off;
    off = (off + bytes + 255) & ~(size_t)255;
    return p;
  };
  _Float16* ln1h  = (_Float16*)alloc((size_t)4096 * 512 * 2);
  _Float16* qkvh  = (_Float16*)alloc((size_t)4096 * 1536 * 2);
  _Float16* attnh = (_Float16*)alloc((size_t)4096 * 512 * 2);
  float*    tmp32 = (float*)   alloc((size_t)4096 * 512 * 4);
  _Float16* qinh  = (_Float16*)alloc((size_t)4096 * 512 * 2);
  _Float16* qh    = (_Float16*)alloc((size_t)4096 * 512 * 2);
  _Float16* xh    = (_Float16*)alloc((size_t)8192 * 512 * 2);
  _Float16* kvh   = (_Float16*)alloc((size_t)8192 * 1024 * 2);
  _Float16* ch    = (_Float16*)alloc((size_t)4096 * 512 * 2);
  (void)ws_size; (void)in_sizes; (void)n_in; (void)out_size;

  // 1) ln1 = LN(q_x) -> f16
  ln_kernel<<<4096, 256, 0, stream>>>(q_x, ln1_g, ln1_b, nullptr, ln1h);
  // 2) qkv = ln1 @ W_qkv -> f16
  gemm_f16_wmma<false, false, false, true>
      <<<dim3(1536 / 128, 4096 / 128), 256, 0, stream>>>(
          ln1h, W_qkv, nullptr, qkvh, nullptr, nullptr, 4096, 1536, 512);
  // 3) chunked local self-attention -> f16 merged heads
  local_attn_kernel<<<512, 256, 0, stream>>>(qkvh, attnh);
  // 4) tmp = attn @ W_ao + b_ao + q_x -> f32
  gemm_f16_wmma<true, true, true, false>
      <<<dim3(512 / 128, 4096 / 128), 256, 0, stream>>>(
          attnh, W_ao, tmp32, nullptr, b_ao, q_x, 4096, 512, 512);
  // 5) q_in = LN2(tmp) -> f32 output slot + f16 for projection
  ln_kernel<<<4096, 256, 0, stream>>>(tmp32, ln2_g, ln2_b, out_qin, qinh);
  // 6) q = q_in @ W_q -> f16
  gemm_f16_wmma<false, false, false, true>
      <<<dim3(512 / 128, 4096 / 128), 256, 0, stream>>>(
          qinh, W_q, nullptr, qh, nullptr, nullptr, 4096, 512, 512);
  // 7) x -> f16 ; kv = x @ W_kv -> f16
  cvt_kernel<<<(8192 * 512) / (256 * 4), 256, 0, stream>>>(x, xh, 8192 * 512);
  gemm_f16_wmma<false, false, false, true>
      <<<dim3(1024 / 128, 8192 / 128), 256, 0, stream>>>(
          xh, W_kv, nullptr, kvh, nullptr, nullptr, 8192, 1024, 512);
  // 8) flash cross-attention -> f16 merged heads
  cross_attn_kernel<<<512, 256, 0, stream>>>(qh, kvh, ch);
  // 9) out = c @ W_o + b_o -> f32 output slot
  gemm_f16_wmma<true, false, true, false>
      <<<dim3(512 / 128, 4096 / 128), 256, 0, stream>>>(
          ch, W_o, out_attn, nullptr, b_o, nullptr, 4096, 512, 512);
}
